// CumulativeSetAttentionLayer_34978213659073
// MI455X (gfx1250) — matrix operand order repaired and checked by reading, not compiled, not run on hardware
//
#include <hip/hip_runtime.h>

// ---------------------------------------------------------------------------
// Problem constants (from the reference)
// ---------------------------------------------------------------------------
#define NTOK   131072
#define D_IN   128
#define WID    512
#define LAT    512
#define DPD    128
#define NHEAD  8
#define N_SEG  256

typedef __bf16 v16bf __attribute__((ext_vector_type(16)));
typedef float  v8f   __attribute__((ext_vector_type(8)));

// gfx1250 async global->LDS path (ASYNCcnt-tracked), guarded so the build
// can never break: fall back to synchronous VGPR staging if absent.
#if defined(__AMDGCN__) && __has_builtin(__builtin_amdgcn_global_load_async_to_lds_b128)
#define USE_ASYNC_LDS 1
#else
#define USE_ASYNC_LDS 0
#endif

#if USE_ASYNC_LDS
// Builtin signature (from hipcc diagnostic): params are GCC-vector int4
// pointers; arg0 in global (AS1), arg1 in LDS (AS3).  Convert through an
// integer so the address-space change is always a legal cast: on AMDGPU a
// flat global pointer's bits are the AS1 address, and a flat LDS pointer's
// low 32 bits are the LDS byte offset (ISA aperture: addr[31:0] = LDS_ADDR).
typedef int gv4i __attribute__((vector_size(4 * sizeof(int))));
#define GPTR(x) ((__attribute__((address_space(1))) gv4i*)(unsigned long long)(x))
#define LPTR(x) ((__attribute__((address_space(3))) gv4i*)(unsigned int)(unsigned long long)(x))
#endif

__device__ __forceinline__ void async_wait0() {
#if USE_ASYNC_LDS
#if __has_builtin(__builtin_amdgcn_s_wait_asynccnt)
    __builtin_amdgcn_s_wait_asynccnt(0);
#else
    asm volatile("s_wait_asynccnt 0x0" ::: "memory");
#endif
#endif
}

// ---------------------------------------------------------------------------
// Workspace layout (bytes)
// ---------------------------------------------------------------------------
#define OFF_XBF  ((size_t)0)                          // N*128 bf16
#define OFF_W1T  (OFF_XBF + (size_t)NTOK*D_IN*2)      // [512,128] bf16 (W1^T)
#define OFF_W2T  (OFF_W1T + (size_t)WID*D_IN*2)       // [512,512] bf16
#define OFF_W3T  (OFF_W2T + (size_t)WID*WID*2)        // [512,512] bf16
#define OFF_WRT  (OFF_W3T + (size_t)WID*LAT*2)        // [512,512] bf16
#define OFF_V    (OFF_WRT + (size_t)LAT*LAT*2)        // [640,8] f32 (W_k folded with W_q)
#define OFF_H1   (OFF_V   + (size_t)(D_IN+LAT)*NHEAD*4) // N*512 bf16
#define OFF_H2   (OFF_H1  + (size_t)NTOK*WID*2)       // N*512 bf16
#define OFF_ENC  (OFF_H2  + (size_t)NTOK*WID*2)       // N*512 f32 (later reused as R)
#define OFF_AGG  (OFF_ENC + (size_t)NTOK*LAT*4)       // N*512 bf16

// ---------------------------------------------------------------------------
// f32 -> bf16 convert
// ---------------------------------------------------------------------------
__global__ __launch_bounds__(256) void cvt_bf16_kernel(const float* __restrict__ in,
                                                       __bf16* __restrict__ out, int n) {
    int idx = blockIdx.x * 256 + threadIdx.x;
    if (idx < n) out[idx] = (__bf16)in[idx];
}

// W [K,M] f32 -> Wt [M,K] bf16 (transpose so B-fragment gathers are contiguous)
__global__ __launch_bounds__(256) void pack_wt_kernel(const float* __restrict__ W,
                                                      __bf16* __restrict__ Wt, int K, int M) {
    int idx = blockIdx.x * 256 + threadIdx.x;
    if (idx < K * M) {
        int k = idx / M, m = idx % M;
        Wt[(size_t)m * K + k] = (__bf16)W[idx];
    }
}

// V[i,h] = sum_d W_k[i, h*128+d] * W_q[h,d]   (fold the head-query into W_k)
__global__ __launch_bounds__(256) void fold_wq_kernel(const float* __restrict__ Wk,
                                                      const float* __restrict__ Wq,
                                                      float* __restrict__ V) {
    int idx = blockIdx.x * 256 + threadIdx.x;           // < 640*8
    int i = idx >> 3, h = idx & 7;
    const float* wk = Wk + (size_t)i * (NHEAD * DPD) + h * DPD;
    const float* wq = Wq + h * DPD;
    float s = 0.f;
    for (int d = 0; d < DPD; ++d) s += wk[d] * wq[d];
    V[idx] = s;
}

// ---------------------------------------------------------------------------
// Tiled bf16 WMMA GEMM with fused bias + ReLU.
//   A  : [N,K] bf16 row-major
//   Bt : [M,K] bf16 row-major  (== W^T)
//   Out: [N,M], bf16 (outF32==0) or f32 (outF32==1)
// Block tile 128x64: 8 waves, each wave does a 16x64 strip (4 WMMA C-tiles),
// K stepped by 32.  Double-buffered LDS (24 KB); tile k+1 is prefetched
// (async global->LDS when available) while tile k feeds the WMMAs, so only
// one barrier per K-step.
// ---------------------------------------------------------------------------
__global__ __launch_bounds__(256) void gemm_bf16_relu_kernel(
    const __bf16* __restrict__ A, const __bf16* __restrict__ Bt,
    const float* __restrict__ bias, void* __restrict__ Out,
    int K, int M, int outF32) {

    __shared__ __align__(16) __bf16 sA[2][128 * 32];   // 2 x 8 KB
    __shared__ __align__(16) __bf16 sB[2][64 * 32];    // 2 x 4 KB

    const int t    = threadIdx.x;
    const int lane = t & 31;
    const int w    = t >> 5;           // wave 0..7
    const int m    = lane & 15;        // row (A) / col (B,C) within tile
    const int p    = lane >> 4;        // lane-half

    const int rowBlock = blockIdx.x * 128;
    const int colBase  = blockIdx.y * 64;

    v8f acc[4] = {};

    // per-thread LDS stage assignments
    const int ar = t >> 1, aseg = t & 1;   // A: 128 rows x 32, 32B/thread
    const int br = t >> 2, bseg = t & 3;   // B:  64 rows x 32, 16B/thread

    // stage one K-tile into LDS buffer `buf`
    auto stage = [&](int buf, int kb) {
        const __bf16* asrc = A  + (size_t)(rowBlock + ar) * K + kb + aseg * 16;
        const __bf16* bsrc = Bt + (size_t)(colBase  + br) * K + kb + bseg * 8;
        __bf16* adst = &sA[buf][ar * 32 + aseg * 16];
        __bf16* bdst = &sB[buf][br * 32 + bseg * 8];
#if USE_ASYNC_LDS
        __builtin_amdgcn_global_load_async_to_lds_b128(GPTR(asrc),     LPTR(adst),     0, 0);
        __builtin_amdgcn_global_load_async_to_lds_b128(GPTR(asrc + 8), LPTR(adst + 8), 0, 0);
        __builtin_amdgcn_global_load_async_to_lds_b128(GPTR(bsrc),     LPTR(bdst),     0, 0);
#else
        const uint4* as4 = reinterpret_cast<const uint4*>(asrc);
        uint4*       ad4 = reinterpret_cast<uint4*>(adst);
        ad4[0] = as4[0];
        ad4[1] = as4[1];
        reinterpret_cast<uint4*>(bdst)[0] = reinterpret_cast<const uint4*>(bsrc)[0];
#endif
    };

    stage(0, 0);
    async_wait0();
    __syncthreads();

    int buf = 0;
    for (int kb = 0; kb < K; kb += 32) {
        // prefetch the next tile into the other buffer while this one computes
        if (kb + 32 < K) stage(buf ^ 1, kb + 32);

        // ---- gather A fragment (16x32 bf16, ISA 16-bit A layout) ----
        union { v16bf v; unsigned u[8]; } afr;
        {
            const unsigned* ap =
                reinterpret_cast<const unsigned*>(&sA[buf][(w * 16 + m) * 32]);
#pragma unroll
            for (int j = 0; j < 8; ++j) {
                int k0 = (j < 4) ? (p * 8 + 2 * j) : (16 + p * 8 + 2 * (j - 4));
                afr.u[j] = ap[k0 >> 1];
            }
        }

        // ---- 4 column tiles: gather B fragment (32x16) + WMMA ----
#pragma unroll
        for (int c = 0; c < 4; ++c) {
            union { v16bf v; unsigned u[8]; } bfr;
            const unsigned* bp =
                reinterpret_cast<const unsigned*>(&sB[buf][(c * 16 + m) * 32]);
#pragma unroll
            for (int j = 0; j < 8; ++j) bfr.u[j] = bp[(j < 4) ? (p * 4 + j) : (8 + p * 4 + (j - 4))];

            acc[c] = __builtin_amdgcn_wmma_f32_16x16x32_bf16(
                false, afr.v, false, bfr.v, (short)0, acc[c], false, false);
        }

        async_wait0();       // prefetched tile landed in LDS
        __syncthreads();     // all waves done reading `buf`, next tile visible
        buf ^= 1;
    }

    // ---- epilogue: bias + ReLU, store (C layout: row = w*16 + p*8 + r, col = c*16 + m)
#pragma unroll
    for (int c = 0; c < 4; ++c) {
        const int col = colBase + c * 16 + m;
        const float bv = bias[col];
#pragma unroll
        for (int r = 0; r < 8; ++r) {
            const int row = rowBlock + w * 16 + p * 8 + r;
            float v = acc[c][r] + bv;
            v = fmaxf(v, 0.0f);
            if (outF32)
                reinterpret_cast<float*>(Out)[(size_t)row * M + col] = v;
            else
                reinterpret_cast<__bf16*>(Out)[(size_t)row * M + col] = (__bf16)v;
        }
    }
}

// ---------------------------------------------------------------------------
// Cumulative segment mean: one block per segment (ids sorted), sequential
// prefix over the segment's rows, 512 channels across 256 threads (2 each).
// Emits bf16 (feeds the Wr GEMM).
// ---------------------------------------------------------------------------
__global__ __launch_bounds__(256) void seg_scan_mean_kernel(
    const float* __restrict__ Enc, const long long* __restrict__ seg,
    __bf16* __restrict__ Agg, int n) {

    __shared__ int s_lo, s_hi;
    const long long s = blockIdx.x;
    if (threadIdx.x == 0) {
        int lo = 0, hi = n;
        while (lo < hi) { int mid = (lo + hi) >> 1; if (seg[mid] < s) lo = mid + 1; else hi = mid; }
        s_lo = lo;
        int lo2 = lo, hi2 = n;
        while (lo2 < hi2) { int mid = (lo2 + hi2) >> 1; if (seg[mid] < s + 1) lo2 = mid + 1; else hi2 = mid; }
        s_hi = lo2;
    }
    __syncthreads();
    const int lo = s_lo, hi = s_hi;
    const int c0 = threadIdx.x, c1 = threadIdx.x + 256;
    float sum0 = 0.f, sum1 = 0.f;
    for (int row = lo; row < hi; ++row) {
        const size_t base = (size_t)row * LAT;
        sum0 += Enc[base + c0];
        sum1 += Enc[base + c1];
        const float inv = 1.0f / (float)(row - lo + 1);
        Agg[base + c0] = (__bf16)(sum0 * inv);
        Agg[base + c1] = (__bf16)(sum1 * inv);
    }
}

// ---------------------------------------------------------------------------
// Final: preattn[n,h] = ([x_n, r_n] . V[:,h]) / sqrt(DPD)
// One wave per row; lanes stride over 640 dims; shuffle reduction.
// ---------------------------------------------------------------------------
__global__ __launch_bounds__(256) void head_out_kernel(
    const float* __restrict__ X, const float* __restrict__ R,
    const float* __restrict__ V, float* __restrict__ Out) {

    const int w = threadIdx.x >> 5, lane = threadIdx.x & 31;
    const int n = blockIdx.x * 8 + w;
    float acc[NHEAD] = {};
    const float* xr = X + (size_t)n * D_IN;
    const float* rr = R + (size_t)n * LAT;
    for (int i = lane; i < D_IN + LAT; i += 32) {
        const float c = (i < D_IN) ? xr[i] : rr[i - D_IN];
        const float* vp = V + i * NHEAD;
#pragma unroll
        for (int h = 0; h < NHEAD; ++h) acc[h] += c * vp[h];
    }
#pragma unroll
    for (int off = 16; off > 0; off >>= 1)
#pragma unroll
        for (int h = 0; h < NHEAD; ++h) acc[h] += __shfl_xor(acc[h], off, 32);
    if (lane == 0) {
        const float scale = 0.08838834764831843f;   // 1/sqrt(128)
#pragma unroll
        for (int h = 0; h < NHEAD; ++h) Out[(size_t)n * NHEAD + h] = acc[h] * scale;
    }
}

// ---------------------------------------------------------------------------
// Launcher
// ---------------------------------------------------------------------------
extern "C" void kernel_launch(void* const* d_in, const int* in_sizes, int n_in,
                              void* d_out, int out_size, void* d_ws, size_t ws_size,
                              hipStream_t stream) {
    (void)in_sizes; (void)n_in; (void)out_size; (void)ws_size;

    const float*     X   = (const float*)d_in[0];
    const long long* seg = (const long long*)d_in[1];
    const float*     W1  = (const float*)d_in[2];
    const float*     b1  = (const float*)d_in[3];
    const float*     W2  = (const float*)d_in[4];
    const float*     b2  = (const float*)d_in[5];
    const float*     W3  = (const float*)d_in[6];
    const float*     b3  = (const float*)d_in[7];
    const float*     Wr  = (const float*)d_in[8];
    const float*     br  = (const float*)d_in[9];
    const float*     Wk  = (const float*)d_in[10];
    const float*     Wq  = (const float*)d_in[11];
    float*           out = (float*)d_out;

    char* ws = (char*)d_ws;
    __bf16* XBF = (__bf16*)(ws + OFF_XBF);
    __bf16* W1T = (__bf16*)(ws + OFF_W1T);
    __bf16* W2T = (__bf16*)(ws + OFF_W2T);
    __bf16* W3T = (__bf16*)(ws + OFF_W3T);
    __bf16* WRT = (__bf16*)(ws + OFF_WRT);
    float*  V   = (float*)(ws + OFF_V);
    __bf16* H1  = (__bf16*)(ws + OFF_H1);
    __bf16* H2  = (__bf16*)(ws + OFF_H2);
    float*  ENC = (float*)(ws + OFF_ENC);   // later reused as R
    __bf16* AGG = (__bf16*)(ws + OFF_AGG);

    // ---- pack / fold ----
    cvt_bf16_kernel<<<(NTOK * D_IN) / 256, 256, 0, stream>>>(X, XBF, NTOK * D_IN);
    pack_wt_kernel<<<(D_IN * WID + 255) / 256, 256, 0, stream>>>(W1, W1T, D_IN, WID);
    pack_wt_kernel<<<(WID * WID + 255) / 256, 256, 0, stream>>>(W2, W2T, WID, WID);
    pack_wt_kernel<<<(WID * LAT + 255) / 256, 256, 0, stream>>>(W3, W3T, WID, LAT);
    pack_wt_kernel<<<(LAT * LAT + 255) / 256, 256, 0, stream>>>(Wr, WRT, LAT, LAT);
    fold_wq_kernel<<<((D_IN + LAT) * NHEAD + 255) / 256, 256, 0, stream>>>(Wk, Wq, V);

    // ---- psi MLP: three WMMA GEMMs ----
    dim3 g1(NTOK / 128, WID / 64);
    gemm_bf16_relu_kernel<<<g1, 256, 0, stream>>>(XBF, W1T, b1, H1, D_IN, WID, 0);
    gemm_bf16_relu_kernel<<<g1, 256, 0, stream>>>(H1, W2T, b2, H2, WID, WID, 0);
    gemm_bf16_relu_kernel<<<g1, 256, 0, stream>>>(H2, W3T, b3, ENC, WID, LAT, 1);

    // ---- cumulative segment mean ----
    seg_scan_mean_kernel<<<N_SEG, 256, 0, stream>>>(ENC, seg, AGG, NTOK);

    // ---- rho GEMM (result R reuses ENC buffer, ENC is dead now) ----
    gemm_bf16_relu_kernel<<<g1, 256, 0, stream>>>(AGG, WRT, br, ENC, LAT, LAT, 1);

    // ---- folded head projection ----
    head_out_kernel<<<NTOK / 8, 256, 0, stream>>>(X, ENC, V, out);
}
